// PCVRHyFormer_43465069035717
// MI455X (gfx1250) — matrix operand order repaired and checked by reading, not compiled, not run on hardware
//
#include <hip/hip_runtime.h>
#include <hip/hip_bf16.h>

// ---------------------------------------------------------------------------
// MI455X (gfx1250): fp16 WMMA GEMM pipeline + flash attention.
//  - Projections: TDM (tensor_load_to_lds) double-buffered weight tiles,
//    s_wait_tensorcnt, WMMA f32_16x16x32_f16 accumulation.
//  - Attention: shared K/V tiles staged via global_load_async_to_lds_b128
//    (ASYNCcnt), online softmax, P transposed through LDS, WMMA PV.
// ---------------------------------------------------------------------------

typedef _Float16     v16h __attribute__((ext_vector_type(16)));
typedef _Float16     v8h  __attribute__((ext_vector_type(8)));
typedef _Float16     h4   __attribute__((ext_vector_type(4)));
typedef float        v8f  __attribute__((ext_vector_type(8)));
typedef unsigned int v4u  __attribute__((ext_vector_type(4)));
typedef unsigned int v8u  __attribute__((ext_vector_type(8)));

#define B_  8
#define L_  1024
#define D_  1024
#define H_  16
#define HD_ 64

// ---- A-operand loader: 16x32 (MxK) fp16 tile, CDNA5 A layout --------------
// lane<16 : M=lane,   elems 0..7 -> K=0..7,   elems 8..15 -> K=16..23
// lane>=16: M=lane-16,elems 0..7 -> K=8..15,  elems 8..15 -> K=24..31
__device__ __forceinline__ v16h load_a16x32(const _Float16* base, int ld, int lane) {
    int row  = lane & 15;
    int koff = (lane >> 4) << 3;               // 0 or 8
    const _Float16* p = base + (size_t)row * ld + koff;
    v8h c0 = *(const v8h*)(p);                 // K = koff .. koff+7
    v8h c1 = *(const v8h*)(p + 16);            // K = koff+16 .. koff+23
    v16h r;
#pragma unroll
    for (int i = 0; i < 8; ++i) { r[i] = c0[i]; r[i + 8] = c1[i]; }
    return r;
}

// ---- B-operand loader: 32x16 (KxN) fp16 tile, CDNA5 B layout --------------
// lane<16 : N=lane, elems = K 0..7,16..23 ; lane>=16: N=lane-16, elems = K 8..15,24..31
// Memory holds the operand as rows of N (e.g. W[N,K] row-major).
__device__ __forceinline__ v16h load_b32x16(const _Float16* base, int ld, int lane) {
    int n  = lane & 15;
    int k0 = (lane >> 4) << 3;                 // 0 or 8
    const _Float16* p = base + (size_t)n * ld + k0;
    v8h c0 = *(const v8h*)(p);
    v8h c1 = *(const v8h*)(p + 16);
    v16h r;
#pragma unroll
    for (int i = 0; i < 8; ++i) { r[i] = c0[i]; r[i + 8] = c1[i]; }
    return r;
}

__device__ __forceinline__ float rmax16(float x) {
#pragma unroll
    for (int m = 1; m <= 8; m <<= 1) x = fmaxf(x, __shfl_xor(x, m));
    return x;
}
__device__ __forceinline__ float rsum16(float x) {
#pragma unroll
    for (int m = 1; m <= 8; m <<= 1) x += __shfl_xor(x, m);
    return x;
}

// ---------------------------------------------------------------------------
// TDM: issue a 2-D tile load (rows x cols fp16) into LDS.
// D# built per CDNA5 ISA ch.8: group0 {count, lds_addr, global_addr, type=2},
// group1 {data_size=2B, tensor_dim0/1, tile_dim0/1, dim0_stride}, g2/g3 = 0.
// ---------------------------------------------------------------------------
__device__ __forceinline__ void tdm_load_2d(const _Float16* gsrc, unsigned lds_byte,
                                            int rows, int cols,
                                            int row_stride_elems, int tensor_rows) {
    unsigned long long ga = (unsigned long long)(const void*)gsrc;
    unsigned dim0 = (unsigned)row_stride_elems;       // tensor dim0 length (elems)
    unsigned dim1 = (unsigned)tensor_rows;
    v4u g0;
    g0[0] = 1u;                                       // count=1 (valid user D#)
    g0[1] = lds_byte;                                 // lds_addr (bytes)
    g0[2] = (unsigned)ga;                             // global_addr[31:0]
    g0[3] = (unsigned)(ga >> 32) | (2u << 30);        // global_addr[56:32] | type=2
    v8u g1;
    g1[0] = (1u << 16);                               // wg_mask=0, data_size=1 (2B)
    g1[1] = (dim0 & 0xFFFFu) << 16;                   // tensor_dim0[15:0]
    g1[2] = (dim0 >> 16) | ((dim1 & 0xFFFFu) << 16);  // dim0 hi | dim1 lo
    g1[3] = (dim1 >> 16) | ((unsigned)cols << 16);    // dim1 hi | tile_dim0
    g1[4] = (unsigned)rows;                           // tile_dim1 (tile_dim2=0)
    g1[5] = dim0;                                     // tensor_dim0_stride[31:0]
    g1[6] = 0u;                                       // stride hi | dim1_stride lo
    g1[7] = 0u;
    v4u gz = {0u, 0u, 0u, 0u};
    asm volatile("tensor_load_to_lds %0, %1, %2, %3"
                 :: "s"(g0), "s"(g1), "s"(gz), "s"(gz) : "memory");
}

// ---------------------------------------------------------------------------
// fp32 -> fp16 conversion (vectorized by 4)
// ---------------------------------------------------------------------------
__global__ __launch_bounds__(256) void cvt_f32_f16(const float* __restrict__ in,
                                                   _Float16* __restrict__ out, int n4) {
    int i = blockIdx.x * blockDim.x + threadIdx.x;
    if (i < n4) {
        float4 v = ((const float4*)in)[i];
        h4 o; o[0] = (_Float16)v.x; o[1] = (_Float16)v.y;
              o[2] = (_Float16)v.z; o[3] = (_Float16)v.w;
        ((h4*)out)[i] = o;
    }
}

// ---------------------------------------------------------------------------
// RoPE (in place, fp16 buffer, fp32 math); Q additionally gets 1/sqrt(HD)
// ---------------------------------------------------------------------------
__global__ __launch_bounds__(256) void rope_kernel(_Float16* __restrict__ x,
                                                   const float* __restrict__ cosb,
                                                   const float* __restrict__ sinb,
                                                   float scale) {
    int t = blockIdx.x * blockDim.x + threadIdx.x;           // B*L*H*32 threads
    if (t >= B_ * L_ * H_ * 32) return;
    int idx = t & 31;
    int h   = (t >> 5) & (H_ - 1);
    int bl  = t >> 9;                                        // b*L + l
    const float* cp = cosb + (size_t)bl * HD_;
    const float* sp = sinb + (size_t)bl * HD_;
    size_t off = (size_t)bl * D_ + h * HD_ + idx;
    float x1 = (float)x[off], x2 = (float)x[off + 32];
    float c1 = cp[idx], s1 = sp[idx];
    float c2 = cp[idx + 32], s2 = sp[idx + 32];
    x[off]      = (_Float16)((x1 * c1 - x2 * s1) * scale);
    x[off + 32] = (_Float16)((x2 * c2 + x1 * s2) * scale);
}

// ---------------------------------------------------------------------------
// Tiled GEMM: out[M,N] = A[M,K] @ W[N,K]^T + bias   (fp16 in, fp32 accum)
// BM=128 (8 waves x 16 rows), BN=64, BK=32. Weight tile streamed by the TDM
// into a double-buffered LDS tile: tile i+1 DMAs while tile i feeds WMMA.
// ---------------------------------------------------------------------------
enum { EPI_F16 = 0, EPI_SIG = 1, EPI_VT = 2, EPI_F32 = 3 };

template <int EPI, bool GATED>
__global__ __launch_bounds__(256) void gemm_kernel(const _Float16* __restrict__ A,
                                                   const _Float16* __restrict__ Gt,
                                                   const _Float16* __restrict__ W,
                                                   const float* __restrict__ bias,
                                                   void* __restrict__ outp,
                                                   int M, int N, int K) {
    __shared__ __align__(16) _Float16 wtile[2][64 * 32];     // 2 x 4 KB
    const int lane  = threadIdx.x & 31;
    const int wave  = threadIdx.x >> 5;
    const int nbase = blockIdx.x * 64;
    const int mbase = blockIdx.y * 128;
    const _Float16* arow = A + (size_t)(mbase + wave * 16) * K;
    const _Float16* grow = GATED ? (Gt + (size_t)(mbase + wave * 16) * K) : nullptr;
    const _Float16* wblk = W + (size_t)nbase * K;            // tile row origin

    const int nsteps = K / 32;
    if (threadIdx.x < 32)                                     // wave 0 drives the TDM
        tdm_load_2d(wblk, (unsigned)(size_t)&wtile[0][0], 64, 32, K, N);

    v8f acc[4] = {};

    for (int i = 0; i < nsteps; ++i) {
        if (threadIdx.x < 32)
            __builtin_amdgcn_s_wait_tensorcnt(0);             // tile i landed
        __syncthreads();                                      // visible to all waves,
                                                              // prior reads retired
        if (threadIdx.x < 32 && (i + 1) < nsteps)             // stream tile i+1
            tdm_load_2d(wblk + (i + 1) * 32,
                        (unsigned)(size_t)&wtile[(i + 1) & 1][0], 64, 32, K, N);

        const int kb = i * 32;
        v16h a = load_a16x32(arow + kb, K, lane);
        if (GATED) {
            v16h g = load_a16x32(grow + kb, K, lane);
            a = a * g;                                        // elementwise gate
        }
        if (kb + 32 < K)
            __builtin_prefetch(arow + kb + 32 + (size_t)(lane & 15) * K, 0, 0);

        const _Float16* wt = &wtile[i & 1][0];
#pragma unroll
        for (int ns = 0; ns < 4; ++ns) {
            v16h bm = load_b32x16(wt + ns * 16 * 32, 32, lane);
            acc[ns] = __builtin_amdgcn_wmma_f32_16x16x32_f16(
                false, a, false, bm, (short)0, acc[ns], false, false);
        }
    }

    // Epilogue: C layout -> memory. row = e + 8*(lane>=16), col = lane&15.
    const int col   = lane & 15;
    const int rbase = (lane >> 4) << 3;
#pragma unroll
    for (int ns = 0; ns < 4; ++ns) {
        int gn = nbase + ns * 16 + col;
        float bv = bias[gn];
#pragma unroll
        for (int e = 0; e < 8; ++e) {
            int gm = mbase + wave * 16 + rbase + e;
            float v = acc[ns][e] + bv;
            if constexpr (EPI == EPI_F16) {
                ((_Float16*)outp)[(size_t)gm * N + gn] = (_Float16)v;
            } else if constexpr (EPI == EPI_SIG) {
                float s = 1.0f / (1.0f + __expf(-v));
                ((_Float16*)outp)[(size_t)gm * N + gn] = (_Float16)s;
            } else if constexpr (EPI == EPI_VT) {
                // write V transposed per head: Vt[b, h, d, pos]
                int bb = gm >> 10, pos = gm & (L_ - 1);
                int hh = gn >> 6,  dd  = gn & (HD_ - 1);
                ((_Float16*)outp)[(((size_t)bb * H_ + hh) * HD_ + dd) * L_ + pos] =
                    (_Float16)v;
            } else {
                ((float*)outp)[(size_t)gm * N + gn] = v;
            }
        }
    }
}

// ---------------------------------------------------------------------------
// Flash attention: 4 waves/WG share one (b,h) 64-row Q block. K/V 64x64 tiles
// staged once per WG via global_load_async_to_lds_b128 (ASYNCcnt), WMMA S and
// PV, online softmax with 16-lane wave32 shuffles.
// ---------------------------------------------------------------------------
__global__ __launch_bounds__(128) void attn_kernel(const _Float16* __restrict__ Q,
                                                   const _Float16* __restrict__ Kh,
                                                   const _Float16* __restrict__ Vt,
                                                   const unsigned char* __restrict__ mask,
                                                   _Float16* __restrict__ O) {
    __shared__ __align__(16) _Float16 ktile[64 * 64];        // 8 KB: key x d
    __shared__ __align__(16) _Float16 vtile[64 * 64];        // 8 KB: d x key
    __shared__ __align__(16) _Float16 pbuf[4][16 * 64];      // 8 KB: per-wave P
    const int lane = threadIdx.x & 31;
    const int wave = threadIdx.x >> 5;
    const int blk  = blockIdx.x;                             // B*H*(L/64)
    const int qt   = blk & 15;
    const int h    = (blk >> 4) & (H_ - 1);
    const int b    = blk >> 8;
    const int qbase = qt * 64 + wave * 16;
    const int col   = lane & 15;
    const int rbase = (lane >> 4) << 3;

    const _Float16* qb = Q + ((size_t)b * L_ + qbase) * D_ + h * HD_;
    v16h qa0 = load_a16x32(qb, D_, lane);                    // d = 0..31
    v16h qa1 = load_a16x32(qb + 32, D_, lane);               // d = 32..63

    const _Float16* kbase_g = Kh + (size_t)b * L_ * D_ + h * HD_;
    const _Float16* vbase_g = Vt + ((size_t)b * H_ + h) * HD_ * L_;

    v8f of[4] = {};
    float m_i[8], l_i[8];
#pragma unroll
    for (int e = 0; e < 8; ++e) { m_i[e] = -1e30f; l_i[e] = 0.0f; }

    for (int kb = 0; kb < L_; kb += 64) {
        __syncthreads();                                     // prior tile reads done
        {
            // 128 threads x 4 chunks x 16B = 8 KB per tile, K and V
            int tid = threadIdx.x;
#pragma unroll
            for (int c = 0; c < 4; ++c) {
                int id = tid * 4 + c;                        // 0..511
                int row = id >> 3, seg = id & 7;
                unsigned kgoff = (unsigned)((((size_t)(kb + row)) * D_ + seg * 8) * 2);
                unsigned kloff = (unsigned)(size_t)&ktile[row * 64 + seg * 8];
                asm volatile("global_load_async_to_lds_b128 %0, %1, %2"
                             :: "v"(kloff), "v"(kgoff), "s"(kbase_g) : "memory");
                unsigned vgoff = (unsigned)((((size_t)row) * L_ + kb + seg * 8) * 2);
                unsigned vloff = (unsigned)(size_t)&vtile[row * 64 + seg * 8];
                asm volatile("global_load_async_to_lds_b128 %0, %1, %2"
                             :: "v"(vloff), "v"(vgoff), "s"(vbase_g) : "memory");
            }
        }
        asm volatile("s_wait_asynccnt 0" ::: "memory");
        __syncthreads();

        v8f s[4];
#pragma unroll
        for (int ns = 0; ns < 4; ++ns) {
            v16h k0 = load_b32x16(&ktile[ns * 16 * 64], 64, lane);       // d 0..31
            v16h k1 = load_b32x16(&ktile[ns * 16 * 64] + 32, 64, lane);  // d 32..63
            v8f acc = {};
            acc = __builtin_amdgcn_wmma_f32_16x16x32_f16(false, qa0, false, k0,
                                                         (short)0, acc, false, false);
            acc = __builtin_amdgcn_wmma_f32_16x16x32_f16(false, qa1, false, k1,
                                                         (short)0, acc, false, false);
            int kpos = kb + ns * 16 + col;
            if (mask[b * L_ + kpos]) {
#pragma unroll
                for (int e = 0; e < 8; ++e) acc[e] = -1e30f;
            }
            s[ns] = acc;
        }
        // online softmax per row (row lives in element e, 16-lane group)
#pragma unroll
        for (int e = 0; e < 8; ++e) {
            float mx = fmaxf(fmaxf(s[0][e], s[1][e]), fmaxf(s[2][e], s[3][e]));
            mx = rmax16(mx);
            float mnew = fmaxf(m_i[e], mx);
            float corr = __expf(m_i[e] - mnew);
            m_i[e] = mnew;
            float rs = 0.0f;
#pragma unroll
            for (int ns = 0; ns < 4; ++ns) {
                float p = __expf(s[ns][e] - mnew);
                s[ns][e] = p;
                rs += p;
            }
            l_i[e] = l_i[e] * corr + rsum16(rs);
#pragma unroll
            for (int ds = 0; ds < 4; ++ds) of[ds][e] *= corr;
        }
        // P: C layout -> per-wave LDS (row major 16x64) -> A layout
#pragma unroll
        for (int ns = 0; ns < 4; ++ns)
#pragma unroll
            for (int e = 0; e < 8; ++e)
                pbuf[wave][(rbase + e) * 64 + ns * 16 + col] = (_Float16)s[ns][e];

        const _Float16* pb = &pbuf[wave][0];
        v16h pa0 = load_a16x32(pb, 64, lane);                // k 0..31
        v16h pa1 = load_a16x32(pb + 32, 64, lane);           // k 32..63
#pragma unroll
        for (int ds = 0; ds < 4; ++ds) {
            const _Float16* vb = &vtile[ds * 16 * 64];
            v16h v0 = load_b32x16(vb, 64, lane);             // k 0..31
            v16h v1 = load_b32x16(vb + 32, 64, lane);        // k 32..63
            of[ds] = __builtin_amdgcn_wmma_f32_16x16x32_f16(false, pa0, false, v0,
                                                            (short)0, of[ds], false, false);
            of[ds] = __builtin_amdgcn_wmma_f32_16x16x32_f16(false, pa1, false, v1,
                                                            (short)0, of[ds], false, false);
        }
    }

    // normalize + write fp16 O[b, pos, h*64 + d]
    float inv[8];
#pragma unroll
    for (int e = 0; e < 8; ++e) inv[e] = (l_i[e] > 0.0f) ? 1.0f / l_i[e] : 0.0f;
#pragma unroll
    for (int ds = 0; ds < 4; ++ds)
#pragma unroll
        for (int e = 0; e < 8; ++e)
            O[((size_t)b * L_ + qbase + rbase + e) * D_ + h * HD_ + ds * 16 + col] =
                (_Float16)(of[ds][e] * inv[e]);
}

// ---------------------------------------------------------------------------
// Host-side orchestration
// ---------------------------------------------------------------------------
extern "C" void kernel_launch(void* const* d_in, const int* in_sizes, int n_in,
                              void* d_out, int out_size, void* d_ws, size_t ws_size,
                              hipStream_t stream) {
    (void)in_sizes; (void)n_in; (void)out_size; (void)ws_size;

    const float* query = (const float*)d_in[0];
    const float* key   = (const float*)d_in[1];
    const float* value = (const float*)d_in[2];
    const float* Wq = (const float*)d_in[3];  const float* bq = (const float*)d_in[4];
    const float* Wk = (const float*)d_in[5];  const float* bk = (const float*)d_in[6];
    const float* Wv = (const float*)d_in[7];  const float* bv = (const float*)d_in[8];
    const float* Wg = (const float*)d_in[9];  const float* bg = (const float*)d_in[10];
    const float* Wo = (const float*)d_in[11]; const float* bo = (const float*)d_in[12];
    const float* rope_cos = (const float*)d_in[13];
    const float* rope_sin = (const float*)d_in[14];
    const unsigned char* maskp = (const unsigned char*)d_in[15];

    const size_t SZ_X = (size_t)B_ * L_ * D_;       // 8,388,608
    const size_t SZ_W = (size_t)D_ * D_;            // 1,048,576

    _Float16* p   = (_Float16*)d_ws;
    _Float16* Xq  = p; p += SZ_X;
    _Float16* Xk  = p; p += SZ_X;
    _Float16* Xv  = p; p += SZ_X;
    _Float16* Wqh = p; p += SZ_W;
    _Float16* Wkh = p; p += SZ_W;
    _Float16* Wvh = p; p += SZ_W;
    _Float16* Wgh = p; p += SZ_W;
    _Float16* Woh = p; p += SZ_W;
    _Float16* Qh  = p; p += SZ_X;
    _Float16* Kh  = p; p += SZ_X;
    _Float16* Vt  = p; p += SZ_X;
    _Float16* Gh  = Xk;   // reuse: Xk dead after K projection
    _Float16* Oh  = Xv;   // reuse: Xv dead after V projection

    // 1) fp32 -> fp16 conversions
    cvt_f32_f16<<<(int)(SZ_X / 4 / 256), 256, 0, stream>>>(query, Xq, (int)(SZ_X / 4));
    cvt_f32_f16<<<(int)(SZ_X / 4 / 256), 256, 0, stream>>>(key,   Xk, (int)(SZ_X / 4));
    cvt_f32_f16<<<(int)(SZ_X / 4 / 256), 256, 0, stream>>>(value, Xv, (int)(SZ_X / 4));
    cvt_f32_f16<<<(int)(SZ_W / 4 / 256), 256, 0, stream>>>(Wq, Wqh, (int)(SZ_W / 4));
    cvt_f32_f16<<<(int)(SZ_W / 4 / 256), 256, 0, stream>>>(Wk, Wkh, (int)(SZ_W / 4));
    cvt_f32_f16<<<(int)(SZ_W / 4 / 256), 256, 0, stream>>>(Wv, Wvh, (int)(SZ_W / 4));
    cvt_f32_f16<<<(int)(SZ_W / 4 / 256), 256, 0, stream>>>(Wg, Wgh, (int)(SZ_W / 4));
    cvt_f32_f16<<<(int)(SZ_W / 4 / 256), 256, 0, stream>>>(Wo, Woh, (int)(SZ_W / 4));

    const int M = B_ * L_, N = D_, K = D_;
    dim3 gg(N / 64, M / 128);   // (16, 64)

    // 2) projections (TDM-streamed weight tiles)
    gemm_kernel<EPI_F16, false><<<gg, 256, 0, stream>>>(Xq, nullptr, Wqh, bq, Qh, M, N, K);
    gemm_kernel<EPI_F16, false><<<gg, 256, 0, stream>>>(Xk, nullptr, Wkh, bk, Kh, M, N, K);
    gemm_kernel<EPI_VT,  false><<<gg, 256, 0, stream>>>(Xv, nullptr, Wvh, bv, Vt, M, N, K);
    gemm_kernel<EPI_SIG, false><<<gg, 256, 0, stream>>>(Xq, nullptr, Wgh, bg, Gh, M, N, K);

    // 3) RoPE (Q gets 1/sqrt(HD) = 0.125 folded in)
    const int ropeT = B_ * L_ * H_ * 32;
    rope_kernel<<<ropeT / 256, 256, 0, stream>>>(Qh, rope_cos, rope_sin, 0.125f);
    rope_kernel<<<ropeT / 256, 256, 0, stream>>>(Kh, rope_cos, rope_sin, 1.0f);

    // 4) flash attention (async-LDS staged K/V)
    attn_kernel<<<B_ * H_ * (L_ / 64), 128, 0, stream>>>(Qh, Kh, Vt, maskp, Oh);

    // 5) gated output projection (fp32 out)
    gemm_kernel<EPI_F32, true><<<gg, 256, 0, stream>>>(Oh, Gh, Woh, bo, d_out, M, N, K);
}
